// CharEmb_7799660609665
// MI455X (gfx1250) — matrix operand order repaired and checked by reading, not compiled, not run on hardware
//
#include <hip/hip_runtime.h>
#include <hip/hip_bf16.h>

// Problem sizes (fixed by the reference)
#define B_ 32
#define S_ 512
#define C_ 32          // chars per word (conv axis length)
#define V_ 101
#define E_ 64
#define F_ 128
#define K_ 3
#define NWORDS (B_*S_) // 16384

// d_ws layout (bf16 = ushort):
//   [0, 24576)            wT[k][f][e]   (3*128*64)  -- conv weights, bf16
//   [24576, 24576+6464)   embT[v][e]    (101*64)    -- embedding table, bf16
#define WS_W_ELEMS (K_*F_*E_)
#define WS_E_OFF   WS_W_ELEMS
#define WS_TOTAL   (WS_W_ELEMS + V_*E_)

typedef __attribute__((ext_vector_type(16))) __bf16 v16bf;
typedef __attribute__((ext_vector_type(8)))  float  v8f;

__device__ __forceinline__ unsigned short f2bf(float f) {
  // round-to-nearest-even fp32 -> bf16 (inputs are finite)
  unsigned u = __float_as_uint(f);
  return (unsigned short)((u + 0x7FFFu + ((u >> 16) & 1u)) >> 16);
}

// ---------------------------------------------------------------------------
// One-time: convert weights (relayout [f][e][k] -> [k][f][e]) and the
// embedding table to bf16 in workspace. Both then live in L2/WGP$.
// ---------------------------------------------------------------------------
__global__ void prep_kernel(const float* __restrict__ emb,
                            const float* __restrict__ w,
                            unsigned short* __restrict__ ws) {
  int idx = blockIdx.x * blockDim.x + threadIdx.x;
  if (idx < WS_W_ELEMS) {
    int e = idx & 63;
    int f = (idx >> 6) & 127;
    int k = idx >> 13;
    ws[idx] = f2bf(w[(f * E_ + e) * K_ + k]);
  }
  int j = idx - WS_W_ELEMS;
  if (j >= 0 && j < V_ * E_) {
    ws[WS_E_OFF + j] = f2bf(emb[j]);
  }
}

// ---------------------------------------------------------------------------
// Main kernel: each wave32 processes whole words.
//   Y[t,f] = sum_{k,e} xT[t+k][e] * wT[k][f][e]   (t in [0,30), padded to 32)
//   out[word][f] = max_t Y[t,f] + bias[f]
// A (16t x 32e) tiles come from LDS (xT built transposed during the embedding
// scatter); B (32e x 16f) tiles come straight from the cached bf16 weights.
// ---------------------------------------------------------------------------
#define WAVES_PER_BLOCK 8
#define XT_ROWS   34          // rows 32,33 are zero pad for the k-shifted reads
#define XT_STRIDE 64          // bf16 per row

__global__ __launch_bounds__(256) void charconv_kernel(
    const int* __restrict__ char_ids,
    const unsigned short* __restrict__ ws,   // bf16 tables (prep_kernel output)
    const float* __restrict__ conv_b,
    float* __restrict__ out) {
  __shared__ __attribute__((aligned(16)))
      unsigned short xT_all[WAVES_PER_BLOCK][XT_ROWS * XT_STRIDE];  // 34816 B

  const int lane = threadIdx.x & 31;
  const int wid  = threadIdx.x >> 5;
  unsigned short* xT = xT_all[wid];

  // zero the two pad rows once (their conv results are masked out of the max,
  // but keep them finite)
  {
    unsigned* z = (unsigned*)(xT + 32 * XT_STRIDE);
    z[lane] = 0u;
    z[lane + 32] = 0u;
  }

  const unsigned short* wT   = ws;              // [3][128][64] bf16
  const unsigned short* embT = ws + WS_E_OFF;   // [101][64]    bf16

  // per-lane bias for each 16-wide filter tile
  float biasr[8];
#pragma unroll
  for (int nt = 0; nt < 8; ++nt) biasr[nt] = conv_b[nt * 16 + (lane & 15)];

  const int rowA   = lane & 15;         // A: lane holds matrix row M = lane&15
  const int klaneA = (lane >> 4) * 8;   // A: K sub-chunk base (0 or 8)
  const int klaneB = (lane >> 4) * 8;   // B: K sub-chunk base (0 or 8)
  const int fB     = lane & 15;         // B: lane holds column N = lane&15

  const int gwave  = blockIdx.x * WAVES_PER_BLOCK + wid;
  const int nwaves = gridDim.x * WAVES_PER_BLOCK;

  for (int word = gwave; word < NWORDS; word += nwaves) {
    // ---- embedding gather + transposed scatter into LDS ------------------
    // lane == char index c'. The raw [C,E] buffer viewed as x[e][c] gives
    // xT[c][e] with c = e'&31, e = 2*c' + (e'>>5): the (e', e'+32) pair of a
    // char's embedding is bf16-adjacent, so each pair packs into one b32
    // store. Conflict-free across lanes.
    {
      int id = char_ids[word * C_ + lane];
      const uint4* er = (const uint4*)(embT + id * E_);  // 8 x uint4 = 64 bf16
      unsigned loa[16], hia[16];
#pragma unroll
      for (int j = 0; j < 4; ++j) {
        uint4 t = er[j];
        loa[4 * j] = t.x; loa[4 * j + 1] = t.y; loa[4 * j + 2] = t.z; loa[4 * j + 3] = t.w;
        uint4 u = er[j + 4];
        hia[4 * j] = u.x; hia[4 * j + 1] = u.y; hia[4 * j + 2] = u.z; hia[4 * j + 3] = u.w;
      }
      unsigned* xTd = (unsigned*)xT;  // row = 32 dwords
#pragma unroll
      for (int p = 0; p < 16; ++p) {
        unsigned a = loa[p], b = hia[p];               // e' = 2p,2p+1 (+32)
        xTd[(2 * p)     * 32 + lane] = (a & 0xFFFFu) | (b << 16);
        xTd[(2 * p + 1) * 32 + lane] = (a >> 16)     | (b & 0xFFFF0000u);
      }
    }
    // same-wave DS ops complete in order: the A-tile loads below see the data

    // ---- 96 x v_wmma_f32_16x16x32_bf16 -----------------------------------
    v8f acc[2][8];
#pragma unroll
    for (int m = 0; m < 2; ++m)
#pragma unroll
      for (int nt = 0; nt < 8; ++nt)
#pragma unroll
        for (int i = 0; i < 8; ++i) acc[m][nt][i] = 0.0f;

    for (int k = 0; k < K_; ++k) {
      for (int c = 0; c < 2; ++c) {                 // e chunk of 32
        v16bf a[2];
#pragma unroll
        for (int m = 0; m < 2; ++m) {               // t tile (t = 16m + rowA)
          const unsigned short* ap =
              xT + (16 * m + rowA + k) * XT_STRIDE + c * 32 + klaneA;
          union { uint4 u[2]; v16bf v; } ua;
          ua.u[0] = *(const uint4*)(ap);            // K = klaneA..+7
          ua.u[1] = *(const uint4*)(ap + 16);       // K = klaneA+16..+23
          a[m] = ua.v;
        }
#pragma unroll
        for (int nt = 0; nt < 8; ++nt) {            // filter tile
          const unsigned short* wp =
              wT + (k * F_ + nt * 16 + fB) * E_ + c * 32 + klaneB;
          union { uint4 u[2]; v16bf v; } ub;
          ub.u[0] = *(const uint4*)(wp);            // K = klaneB..+7
          ub.u[1] = *(const uint4*)(wp + 16);       // K = klaneB+16..+23
          acc[0][nt] = __builtin_amdgcn_wmma_f32_16x16x32_bf16(
              false, a[0], false, ub.v, (short)0, acc[0][nt], false, false);
          acc[1][nt] = __builtin_amdgcn_wmma_f32_16x16x32_bf16(
              false, a[1], false, ub.v, (short)0, acc[1][nt], false, false);
        }
      }
    }

    // ---- max over t, add bias, store -------------------------------------
    // D layout: VGPR r, lane l -> M = r + 8*(l>>4), N = l&15.
    // Tile m=1: t = 16 + M; t=30,31 (invalid) are r=6,7 on lanes>=16 only.
    float* op = out + word * F_;
#pragma unroll
    for (int nt = 0; nt < 8; ++nt) {
      float mx = acc[0][nt][0];
#pragma unroll
      for (int r = 1; r < 8; ++r) mx = fmaxf(mx, acc[0][nt][r]);
#pragma unroll
      for (int r = 0; r < 6; ++r) mx = fmaxf(mx, acc[1][nt][r]);
      if (lane < 16) {  // these lanes' r=6,7 are t=22,23: valid
        mx = fmaxf(mx, acc[1][nt][6]);
        mx = fmaxf(mx, acc[1][nt][7]);
      }
      mx = fmaxf(mx, __shfl_xor(mx, 16, 32));  // merge M halves (lane ^ 16)
      if (lane < 16) op[nt * 16 + lane] = mx + biasr[nt];
    }
  }
}

// ---------------------------------------------------------------------------
extern "C" void kernel_launch(void* const* d_in, const int* in_sizes, int n_in,
                              void* d_out, int out_size, void* d_ws, size_t ws_size,
                              hipStream_t stream) {
  (void)in_sizes; (void)n_in; (void)out_size; (void)ws_size;
  const int*   char_ids = (const int*)d_in[0];
  const float* emb      = (const float*)d_in[1];
  const float* conv_w   = (const float*)d_in[2];
  const float* conv_b   = (const float*)d_in[3];
  unsigned short* ws    = (unsigned short*)d_ws;   // needs 62,080 B of scratch
  float* out            = (float*)d_out;

  prep_kernel<<<(WS_TOTAL + 255) / 256, 256, 0, stream>>>(emb, conv_w, ws);
  // 512 blocks x 8 waves = 4096 waves -> 4 words per wave
  charconv_kernel<<<512, 256, 0, stream>>>(char_ids, ws, conv_b, out);
}
